// AdaptiveProductHead_58686433132713
// MI455X (gfx1250) — matrix operand order repaired and checked by reading, not compiled, not run on hardware
//
#include <hip/hip_runtime.h>
#include <hip/hip_bf16.h>

typedef __attribute__((ext_vector_type(16))) _Float16 v16h;
typedef __attribute__((ext_vector_type(8)))  _Float16 v8h;
typedef __attribute__((ext_vector_type(8)))  float    v8f;

#define DIMD   768
#define KSTEPS (DIMD/32)      // 24 WMMA K-steps for the projection GEMM
#define EPSV   1e-7f

// ---- WMMA operand element maps (per CDNA5 ISA §7.12.2, wave32) ----
// B matrix (32x16, f16, 8 VGPRs = 16 halves/lane): lane L holds N = L%16; base = (L<16 ? 0 : 8);
// halves 0..7 -> K = base+i, halves 8..15 -> K = 16+base+(i-8).
static __device__ __forceinline__ int b_kmap(int lane, int i) {
    int base = (lane >> 4) << 3;
    return (i < 8) ? (base + i) : (16 + base + (i - 8));
}
// A matrix (16x32, f16): lane L holds M = L%16; base = (L<16 ? 0 : 8);
// halves 0..7 -> K = base+i, halves 8..15 -> K = 16+base+(i-8).
static __device__ __forceinline__ int a_kmap(int lane, int i) {
    int base = (lane >> 4) << 3;
    return (i < 8) ? (base + i) : (16 + base + (i - 8));
}

// ------------------------------------------------------------------
// Kernel 0: pack Wcat = [We(32) ; Wh(16) ; Ws(16)] (64 x 768, f32)
// into WMMA B-operand f16 layout: [ntile(4)][kstep(24)][lane(32)][half(16)]
// ------------------------------------------------------------------
__global__ void pack_w_kernel(const float* __restrict__ We,
                              const float* __restrict__ Wh,
                              const float* __restrict__ Ws,
                              _Float16* __restrict__ wpack) {
    int idx = blockIdx.x * blockDim.x + threadIdx.x;
    if (idx >= 4 * KSTEPS * 32 * 16) return;
    int i  = idx & 15;
    int L  = (idx >> 4) & 31;
    int kk = (idx >> 9) % KSTEPS;
    int n  = idx / (KSTEPS * 512);
    int K    = kk * 32 + b_kmap(L, i);
    int ncol = n * 16 + (L & 15);
    float v;
    if (ncol < 32)      v = We[ncol * DIMD + K];
    else if (ncol < 48) v = Wh[(ncol - 32) * DIMD + K];
    else                v = Ws[(ncol - 48) * DIMD + K];
    wpack[idx] = (_Float16)v;
}

// ------------------------------------------------------------------
// Kernel 1: project 16 rows of X through Wcat via WMMA (K=768, 24 steps),
// apply normalize / expmap0 / normalize, then emit packed f16 operand
// tiles for the retrieval phase.
//   mode 0 (corpus): emit B-format tiles  [tile][mat(3)][lane][16 halves]
//   mode 1 (query):  emit A-format tiles  (same byte layout, A element map)
// mats: 0 = e (K=32), 1 = h (K 0..15 real, 16..31 zero), 2 = s (same pad).
// Also writes ||h||^2 per row (yn for corpus, xn for queries).
// ------------------------------------------------------------------
__global__ __launch_bounds__(128) void project_kernel(
    const float* __restrict__ x, const _Float16* __restrict__ wpack,
    const float* __restrict__ be, const float* __restrict__ bh,
    const float* __restrict__ bs, const float* __restrict__ scale_h_p,
    _Float16* __restrict__ pack_out, float* __restrict__ hnorm_out, int mode)
{
    __shared__ _Float16 ldsX[16 * DIMD];   // 24 KB: 16-row f16 tile of X
    __shared__ float    feat[16 * 64];     // 4 KB: f32 GEMM result

    const int tid  = threadIdx.x;
    const int lane = tid & 31;
    const int wv   = tid >> 5;             // wave = N-tile (0..3)
    const int tile = blockIdx.x;
    const long row0 = (long)tile * 16;

    // Cooperative load of the 16x768 X tile, f32 -> f16 (coalesced: rows contiguous)
    for (int idx = tid; idx < 16 * DIMD; idx += 128)
        ldsX[idx] = (_Float16)x[row0 * DIMD + idx];
    __syncthreads();

    // --- WMMA GEMM: feat[16x64] = Xtile @ Wcat^T ---
    v8f acc = {};
    const int abase = (lane >> 4) << 3;    // A-map K base (0 or 8)
    const v16h* wp = (const v16h*)wpack;
    const int arow = (lane & 15) * DIMD;
    for (int kk = 0; kk < KSTEPS; ++kk) {
        // A fragment: two contiguous 16B LDS reads per lane (ds_load_b128)
        v8h alo = *(const v8h*)&ldsX[arow + kk * 32 + abase];
        v8h ahi = *(const v8h*)&ldsX[arow + kk * 32 + 16 + abase];
        v16h a;
        #pragma unroll
        for (int i = 0; i < 8; ++i) { a[i] = alo[i]; a[i + 8] = ahi[i]; }
        // B fragment: 32B coalesced global load (L2-resident, 96 KB total)
        v16h b = wp[(wv * KSTEPS + kk) * 32 + lane];
        acc = __builtin_amdgcn_wmma_f32_16x16x32_f16(
                  false, a, false, b, (short)0, acc, false, false);
    }
    // Scatter accumulator into feat[row][col] (C layout: VGPR v -> M = v | v+8)
    {
        int rhi  = (lane >> 4) << 3;
        int coln = wv * 16 + (lane & 15);
        #pragma unroll
        for (int v = 0; v < 8; ++v)
            feat[(v + rhi) * 64 + coln] = acc[v];
    }
    __syncthreads();

    // --- Per-row nonlinear transforms (16 rows, one thread each) ---
    const float sc = *scale_h_p;
    if (tid < 16) {
        float* f = &feat[tid * 64];
        // e: bias + L2 normalize (dims 0..31)
        float ne = 0.f;
        for (int c = 0; c < 32; ++c) { float t = f[c] + be[c]; f[c] = t; ne += t * t; }
        float inve = 1.0f / sqrtf(fmaxf(ne, 1e-30f));
        for (int c = 0; c < 32; ++c) f[c] *= inve;
        // h: (bias, * scale_h), expmap0 with c=1: tanh(n)*u/n  (dims 32..47)
        float nh2 = 0.f;
        for (int c = 0; c < 16; ++c) { float t = (f[32 + c] + bh[c]) * sc; f[32 + c] = t; nh2 += t * t; }
        float nh  = fmaxf(sqrtf(nh2), 1e-15f);
        float th  = tanhf(nh);
        float fac = th / nh;
        for (int c = 0; c < 16; ++c) f[32 + c] *= fac;
        hnorm_out[tile * 16 + tid] = th * th;     // ||expmap0||^2
        // s: bias + L2 normalize (dims 48..63)
        float ns = 0.f;
        for (int c = 0; c < 16; ++c) { float t = f[48 + c] + bs[c]; f[48 + c] = t; ns += t * t; }
        float invs = 1.0f / sqrtf(fmaxf(ns, 1e-30f));
        for (int c = 0; c < 16; ++c) f[48 + c] *= invs;
    }
    __syncthreads();

    // --- Emit packed operand tile: [mat(3)][lane(32)][half(16)] f16, coalesced ---
    _Float16* dst = pack_out + (long)tile * 3 * 512;
    for (int idx = tid; idx < 3 * 512; idx += 128) {
        int m = idx >> 9;
        int L = (idx >> 4) & 31;
        int i = idx & 15;
        int r = L & 15;
        int K = mode ? a_kmap(L, i) : b_kmap(L, i);
        float v;
        if (m == 0)       v = feat[r * 64 + K];                         // e, K<32
        else {
            int o = (m == 1) ? 32 : 48;                                 // h or s
            v = (K < 16) ? feat[r * 64 + o + K] : 0.0f;                 // zero-pad K 16..31
        }
        dst[idx] = (_Float16)v;
    }
}

// ------------------------------------------------------------------
// Kernel 2: w = softplus(relu(x_q @ W1^T + b1) @ W2^T + b2)  -> [nq,3]
// One wave per query row; lane t owns hidden unit t; wave-shuffle reduce.
// ------------------------------------------------------------------
__global__ __launch_bounds__(32) void weights_kernel(
    const float* __restrict__ xq, const float* __restrict__ W1,
    const float* __restrict__ b1, const float* __restrict__ W2,
    const float* __restrict__ b2, float* __restrict__ wq)
{
    int row = blockIdx.x;
    int t   = threadIdx.x;          // 0..31
    float acc = b1[t];
    const float* xr = xq + (long)row * DIMD;
    const float* w1 = W1 + (long)t * DIMD;
    for (int k = 0; k < DIMD; ++k) acc += xr[k] * w1[k];
    float h = fmaxf(acc, 0.0f);
    #pragma unroll
    for (int j = 0; j < 3; ++j) {
        float p = h * W2[j * 32 + t];
        #pragma unroll
        for (int off = 16; off >= 1; off >>= 1) p += __shfl_xor(p, off, 32);
        if (t == 0) wq[row * 3 + j] = log1pf(expf(p + b2[j]));   // softplus
    }
}

// ------------------------------------------------------------------
// Kernel 3: main retrieval. 8 waves/block; each wave owns one 16q x 16c tile:
// 3x v_wmma_f32_16x16x32_f16 (e / h / s dots), then the distance epilogue.
// ------------------------------------------------------------------
__global__ __launch_bounds__(256) void retrieve_kernel(
    const _Float16* __restrict__ qpack, const _Float16* __restrict__ cpack,
    const float* __restrict__ xn_q, const float* __restrict__ yn_c,
    const float* __restrict__ wq, float* __restrict__ out, int nc)
{
    const int lane  = threadIdx.x & 31;
    const int wv    = threadIdx.x >> 5;
    const int ctile = blockIdx.x * 8 + wv;
    const int qtile = blockIdx.y;

    const v16h* qp = (const v16h*)qpack;
    const v16h* cp = (const v16h*)cpack;
    v16h a0 = qp[((long)qtile * 3 + 0) * 32 + lane];
    v16h a1 = qp[((long)qtile * 3 + 1) * 32 + lane];
    v16h a2 = qp[((long)qtile * 3 + 2) * 32 + lane];
    v16h b0 = cp[((long)ctile * 3 + 0) * 32 + lane];
    v16h b1 = cp[((long)ctile * 3 + 1) * 32 + lane];
    v16h b2 = cp[((long)ctile * 3 + 2) * 32 + lane];

    v8f de = {}, dh = {}, ds = {};
    de = __builtin_amdgcn_wmma_f32_16x16x32_f16(false, a0, false, b0, (short)0, de, false, false);
    dh = __builtin_amdgcn_wmma_f32_16x16x32_f16(false, a1, false, b1, (short)0, dh, false, false);
    ds = __builtin_amdgcn_wmma_f32_16x16x32_f16(false, a2, false, b2, (short)0, ds, false, false);

    const int col = ctile * 16 + (lane & 15);
    const float yn = yn_c[col];
    const int rhi = (lane >> 4) << 3;       // C layout: VGPR v -> row v | v+8
    #pragma unroll
    for (int v = 0; v < 8; ++v) {
        int q = qtile * 16 + v + rhi;
        float w0 = wq[q * 3 + 0], w1 = wq[q * 3 + 1], w2 = wq[q * 3 + 2];
        float xn = xn_q[q];
        float dote = de[v], doth = dh[v], dots = ds[v];
        // Euclidean (unit vectors)
        float dist_e = 2.0f - 2.0f * dote;
        // Spherical
        float dc = fminf(fmaxf(dots, -1.0f + EPSV), 1.0f - EPSV);
        float ac = acosf(dc);
        float dist_s = ac * ac;
        // Hyperbolic (Poincare, c = 1)
        float alpha = 1.0f - 2.0f * doth + yn;
        float beta  = 1.0f - xn;
        float num_sq = alpha * alpha * xn - 2.0f * alpha * beta * doth + beta * beta * yn;
        float den = fmaxf(1.0f - 2.0f * doth + xn * yn, 1e-15f);
        float t = sqrtf(fmaxf(num_sq, 0.0f)) / den;
        t = fminf(fmaxf(t, 0.0f), 1.0f - EPSV);
        float dg = 2.0f * atanhf(t);
        float dist_h = dg * dg;
        out[(long)q * nc + col] = -(w0 * dist_e + w1 * dist_h + w2 * dist_s);
    }
}

// ------------------------------------------------------------------
extern "C" void kernel_launch(void* const* d_in, const int* in_sizes, int n_in,
                              void* d_out, int out_size, void* d_ws, size_t ws_size,
                              hipStream_t stream) {
    const float* x_q     = (const float*)d_in[0];
    const float* x_c     = (const float*)d_in[1];
    const float* We      = (const float*)d_in[2];
    const float* be      = (const float*)d_in[3];
    const float* Wh      = (const float*)d_in[4];
    const float* bh      = (const float*)d_in[5];
    const float* Ws      = (const float*)d_in[6];
    const float* bs      = (const float*)d_in[7];
    const float* scale_h = (const float*)d_in[8];
    const float* W1      = (const float*)d_in[9];
    const float* b1      = (const float*)d_in[10];
    const float* W2      = (const float*)d_in[11];
    const float* b2      = (const float*)d_in[12];
    float* out = (float*)d_out;

    const int nq = in_sizes[0] / DIMD;   // 512
    const int nc = in_sizes[1] / DIMD;   // 65536

    // Workspace carve-up (all offsets 32B-aligned)
    char* ws = (char*)d_ws;
    _Float16* wpack = (_Float16*)ws;                                   // 98304 B
    size_t off = (size_t)4 * KSTEPS * 512 * sizeof(_Float16);
    _Float16* qpack = (_Float16*)(ws + off);
    off += (size_t)(nq / 16) * 3 * 512 * sizeof(_Float16);             // 98304 B
    _Float16* cpack = (_Float16*)(ws + off);
    off += (size_t)(nc / 16) * 3 * 512 * sizeof(_Float16);             // ~12.6 MB
    float* xn_q = (float*)(ws + off);  off += (size_t)nq * sizeof(float);
    float* yn_c = (float*)(ws + off);  off += (size_t)nc * sizeof(float);
    float* wqb  = (float*)(ws + off);

    pack_w_kernel<<<(4 * KSTEPS * 512 + 255) / 256, 256, 0, stream>>>(We, Wh, Ws, wpack);
    project_kernel<<<nc / 16, 128, 0, stream>>>(x_c, wpack, be, bh, bs, scale_h, cpack, yn_c, 0);
    project_kernel<<<nq / 16, 128, 0, stream>>>(x_q, wpack, be, bh, bs, scale_h, qpack, xn_q, 1);
    weights_kernel<<<nq, 32, 0, stream>>>(x_q, W1, b1, W2, b2, wqb);
    dim3 grid(nc / 128, nq / 16);
    retrieve_kernel<<<grid, 256, 0, stream>>>(qpack, cpack, xn_q, yn_c, wqb, out, nc);
}
